// LSWTTokenPoolerAttention_12773232738466
// MI455X (gfx1250) — matrix operand order repaired and checked by reading, not compiled, not run on hardware
//
#include <hip/hip_runtime.h>

#define BATCH 2
#define SEQ 1024
#define DMODEL 1024
#define NH 16
#define DKH 64
#define NTOK (BATCH * SEQ)

typedef __attribute__((ext_vector_type(16))) __bf16 v16bf;
typedef __attribute__((ext_vector_type(8)))  float  v8f;

union Frag16 { v16bf v; uint4 u[2]; };
static_assert(sizeof(Frag16) == 32, "frag size");

__device__ __forceinline__ unsigned short f2bf(float f) {
    unsigned int u = __float_as_uint(f);
    u += 0x7FFFu + ((u >> 16) & 1u);     // round-to-nearest-even
    return (unsigned short)(u >> 16);
}
__device__ __forceinline__ float bf2f(unsigned short h) {
    return __uint_as_float(((unsigned int)h) << 16);
}

// ---------------------------------------------------------------------------
// Token metadata: segment ids (cumsum of SEP==1), flags bit0=valid, bit1=class
// ---------------------------------------------------------------------------
__global__ void meta_kernel(const int* __restrict__ ids,
                            int* __restrict__ seg, int* __restrict__ flg) {
    __shared__ int sc[SEQ];
    int b = blockIdx.x, t = threadIdx.x;
    int id = ids[b * SEQ + t];
    sc[t] = (id == 1) ? 1 : 0;
    __syncthreads();
    for (int off = 1; off < SEQ; off <<= 1) {
        int v = (t >= off) ? sc[t - off] : 0;
        __syncthreads();
        sc[t] += v;
        __syncthreads();
    }
    seg[b * SEQ + t] = sc[t];
    flg[b * SEQ + t] = ((id != 0) ? 1 : 0) | ((id == 2) ? 2 : 0);
}

// ---------------------------------------------------------------------------
// fp32 -> bf16 conversion
// ---------------------------------------------------------------------------
__global__ void convert_kernel(const float* __restrict__ src,
                               unsigned short* __restrict__ dst, int n) {
    int i = blockIdx.x * blockDim.x + threadIdx.x;
    if (i < n) dst[i] = f2bf(src[i]);
}

// ---------------------------------------------------------------------------
// LayerNorm: one block (256 thr) per token, fp32 in -> bf16 out
// ---------------------------------------------------------------------------
__global__ __launch_bounds__(256) void ln_kernel(const float* __restrict__ x,
                                                 const float* __restrict__ w,
                                                 const float* __restrict__ bb,
                                                 unsigned short* __restrict__ y) {
    int tok = blockIdx.x, t = threadIdx.x;
    float4 xv = ((const float4*)(x + (size_t)tok * DMODEL))[t];
    float s  = xv.x + xv.y + xv.z + xv.w;
    float sq = xv.x * xv.x + xv.y * xv.y + xv.z * xv.z + xv.w * xv.w;
    #pragma unroll
    for (int m = 1; m < 32; m <<= 1) {
        s  += __shfl_xor(s,  m, 32);
        sq += __shfl_xor(sq, m, 32);
    }
    __shared__ float ps[8], psq[8];
    int wid = t >> 5, lane = t & 31;
    if (lane == 0) { ps[wid] = s; psq[wid] = sq; }
    __syncthreads();
    float ts = 0.f, tsq = 0.f;
    #pragma unroll
    for (int i = 0; i < 8; ++i) { ts += ps[i]; tsq += psq[i]; }
    float mean = ts * (1.0f / DMODEL);
    float var  = tsq * (1.0f / DMODEL) - mean * mean;
    float rstd = rsqrtf(var + 1e-5f);
    int d = t * 4;
    ushort4 o;
    o.x = f2bf((xv.x - mean) * rstd * w[d + 0] + bb[d + 0]);
    o.y = f2bf((xv.y - mean) * rstd * w[d + 1] + bb[d + 1]);
    o.z = f2bf((xv.z - mean) * rstd * w[d + 2] + bb[d + 2]);
    o.w = f2bf((xv.w - mean) * rstd * w[d + 3] + bb[d + 3]);
    *(ushort4*)&y[(size_t)tok * DMODEL + d] = o;
}

// ---------------------------------------------------------------------------
// bf16 GEMM: out[tok, o] = sum_k A[tok,k] * W[o,k]  (+bias)
// Block tile 128x64, 8 waves (4x2), each wave 2x2 WMMA 16x16 tiles, K-step 32.
// Global->LDS staging uses CDNA5 async-to-LDS DMA (ASYNCcnt) with ping-pong
// double buffering: tile t+1 copies overlap tile t WMMAs.
// mode 1: bf16 row-major; mode 2: bf16 V-transposed [b][h][dk][s];
// mode 3: fp32 residual accumulate in-place (outf += acc + bias)
// ---------------------------------------------------------------------------
__global__ __launch_bounds__(256) void gemm_kernel(const unsigned short* __restrict__ A,
                                                   const unsigned short* __restrict__ W,
                                                   const float* __restrict__ bias,
                                                   unsigned short* __restrict__ outbf,
                                                   float* __restrict__ outf,
                                                   int mode) {
    __shared__ unsigned short As[2][128 * 40];  // rows padded to 80B
    __shared__ unsigned short Bs[2][64 * 40];
    const int K = DMODEL, NO = DMODEL;
    int tid = threadIdx.x;
    int nBase = blockIdx.x * 64;
    int mBase = blockIdx.y * 128;
    int wid = tid >> 5, lane = tid & 31;
    int half = lane >> 4, l16 = lane & 15;
    int wm = wid & 3, wn = wid >> 2;

    // per-thread 16B staging chunks: A needs 512 chunks (2/thread), B 256 (1/thread)
    const int rowA0 = tid >> 2;          // 0..63
    const int rowA1 = (tid >> 2) + 64;   // 64..127
    const int rowB  = tid >> 2;
    const int kc    = (tid & 3) * 8;     // u16 offset within 32-wide K slab

    // issue one tile's global->LDS async DMA (3 async ops per wave)
    auto issue_tile = [&](int buf, int k0) {
        unsigned la0 = (unsigned)(unsigned long long)(uintptr_t)&As[buf][rowA0 * 40 + kc];
        unsigned la1 = (unsigned)(unsigned long long)(uintptr_t)&As[buf][rowA1 * 40 + kc];
        unsigned lb  = (unsigned)(unsigned long long)(uintptr_t)&Bs[buf][rowB  * 40 + kc];
        unsigned long long ga0 =
            (unsigned long long)(uintptr_t)(A + (size_t)(mBase + rowA0) * K + k0 + kc);
        unsigned long long ga1 =
            (unsigned long long)(uintptr_t)(A + (size_t)(mBase + rowA1) * K + k0 + kc);
        unsigned long long gb =
            (unsigned long long)(uintptr_t)(W + (size_t)(nBase + rowB) * K + k0 + kc);
        asm volatile("global_load_async_to_lds_b128 %0, %1, off"
                     :: "v"(la0), "v"(ga0) : "memory");
        asm volatile("global_load_async_to_lds_b128 %0, %1, off"
                     :: "v"(la1), "v"(ga1) : "memory");
        asm volatile("global_load_async_to_lds_b128 %0, %1, off"
                     :: "v"(lb), "v"(gb) : "memory");
    };

    v8f acc[2][2];
    const v8f VZERO = {0.f, 0.f, 0.f, 0.f, 0.f, 0.f, 0.f, 0.f};
    #pragma unroll
    for (int i = 0; i < 2; ++i)
        #pragma unroll
        for (int j = 0; j < 2; ++j) acc[i][j] = VZERO;

    issue_tile(0, 0);
    int t = 0;
    for (int k0 = 0; k0 < K; k0 += 32, t ^= 1) {
        if (k0 + 32 < K) {
            issue_tile(t ^ 1, k0 + 32);     // overlap next tile DMA with compute
            if (k0 + 64 < K) {              // warm L2 one tile further ahead
                __builtin_prefetch(A + (size_t)(mBase + rowA0) * K + k0 + 64 + kc, 0, 1);
                __builtin_prefetch(W + (size_t)(nBase + rowB) * K + k0 + 64 + kc, 0, 1);
            }
            // async loads retire in order: <=3 outstanding => current tile landed
            asm volatile("s_wait_asynccnt 3" ::: "memory");
        } else {
            asm volatile("s_wait_asynccnt 0" ::: "memory");
        }
        __syncthreads();

        Frag16 af[2], bfg[2];
        #pragma unroll
        for (int i = 0; i < 2; ++i) {   // A frag: K interleave {0-7,16-23}/{8-15,24-31}
            int row = wm * 32 + i * 16 + l16;
            af[i].u[0] = *(const uint4*)&As[t][row * 40 + (half ? 8 : 0)];
            af[i].u[1] = *(const uint4*)&As[t][row * 40 + (half ? 24 : 16)];
        }
        #pragma unroll
        for (int j = 0; j < 2; ++j) {   // B frag: K interleave {0-7,16-23}/{8-15,24-31}
            int row = wn * 32 + j * 16 + l16;
            bfg[j].u[0] = *(const uint4*)&Bs[t][row * 40 + half * 8];
            bfg[j].u[1] = *(const uint4*)&Bs[t][row * 40 + half * 8 + 16];
        }
        #pragma unroll
        for (int i = 0; i < 2; ++i)
            #pragma unroll
            for (int j = 0; j < 2; ++j)
                acc[i][j] = __builtin_amdgcn_wmma_f32_16x16x32_bf16(
                    false, af[i].v, false, bfg[j].v, (short)0, acc[i][j], false, false);
        __syncthreads();   // protect buffer t before it is overwritten next round
    }

    #pragma unroll
    for (int i = 0; i < 2; ++i) {
        #pragma unroll
        for (int j = 0; j < 2; ++j) {
            #pragma unroll
            for (int v = 0; v < 8; ++v) {
                int r  = mBase + wm * 32 + i * 16 + v + half * 8;
                int cc = nBase + wn * 32 + j * 16 + l16;
                float val = acc[i][j][v] + (bias ? bias[cc] : 0.0f);
                if (mode == 1) {
                    outbf[(size_t)r * NO + cc] = f2bf(val);
                } else if (mode == 2) {
                    int bb = r >> 10, ss = r & 1023;
                    int hh = cc >> 6, dd = cc & 63;
                    outbf[(size_t)((bb * NH + hh) * DKH + dd) * SEQ + ss] = f2bf(val);
                } else {
                    size_t idx = (size_t)r * NO + cc;
                    outf[idx] = outf[idx] + val;
                }
            }
        }
    }
}

// ---------------------------------------------------------------------------
// Per-head dot: out[tok,h] = act(ln[tok,:] . W[h,:] + b[h]); one wave/output
// ---------------------------------------------------------------------------
__global__ __launch_bounds__(256) void hdot_kernel(const unsigned short* __restrict__ A,
                                                   const unsigned short* __restrict__ W,
                                                   const float* __restrict__ bias,
                                                   float* __restrict__ out,
                                                   int sig) {
    int wid = threadIdx.x >> 5, lane = threadIdx.x & 31;
    int job = blockIdx.x * 8 + wid;
    int tok = job >> 4, h = job & 15;
    const unsigned short* a = A + (size_t)tok * DMODEL;
    const unsigned short* w = W + (size_t)h * DMODEL;
    float sum = 0.f;
    #pragma unroll
    for (int c = 0; c < 4; ++c) {
        int d = c * 256 + lane * 8;
        uint4 av = *(const uint4*)(a + d);
        uint4 wv = *(const uint4*)(w + d);
        const unsigned short* ap = (const unsigned short*)&av;
        const unsigned short* wp = (const unsigned short*)&wv;
        #pragma unroll
        for (int e = 0; e < 8; ++e) sum += bf2f(ap[e]) * bf2f(wp[e]);
    }
    #pragma unroll
    for (int m = 1; m < 32; m <<= 1) sum += __shfl_xor(sum, m, 32);
    if (lane == 0) {
        if (bias) sum += bias[h];
        if (sig) sum = 1.0f / (1.0f + expf(-sum));
        out[(size_t)tok * NH + h] = sum;
    }
}

// ---------------------------------------------------------------------------
// Flash attention. One wave = one (b,h,16-query tile); 32-key chunks.
// MODE 0: self sdpa mask; MODE 1: cross sdpa mask; MODE 2: pool (aproj scores,
// self mask). V consumed from dk-major transposed layout Vt[b][h][dk][s].
// ---------------------------------------------------------------------------
template <int MODE>
__global__ __launch_bounds__(128) void attn_kernel(const unsigned short* __restrict__ qbf,
                                                   const unsigned short* __restrict__ kbf,
                                                   const unsigned short* __restrict__ vt,
                                                   const float* __restrict__ aproj,
                                                   const float* __restrict__ gate,
                                                   const int* __restrict__ seg,
                                                   const int* __restrict__ flg,
                                                   unsigned short* __restrict__ obf) {
    __shared__ unsigned short Pb[4][16 * 40];
    int wid = threadIdx.x >> 5, lane = threadIdx.x & 31;
    int half = lane >> 4, l16 = lane & 15;
    int job = blockIdx.x * 4 + wid;              // [0, B*NH*(SEQ/16))
    int b = job >> 10;
    int h = (job >> 6) & 15;
    int q0 = (job & 63) * 16;
    int qhi = q0 + 15;
    float hs = exp2f(-(float)(h + 1) * 0.5f);    // exp2(-(h+1)*SLOPE/H)
    const float NEGI = -1e30f;
    const v8f VZERO = {0.f, 0.f, 0.f, 0.f, 0.f, 0.f, 0.f, 0.f};

    Frag16 qf[2];
    if constexpr (MODE != 2) {
        size_t qbase = ((size_t)(b * SEQ + (q0 + l16))) * DMODEL + h * DKH;
        #pragma unroll
        for (int c = 0; c < 2; ++c) {
            qf[c].u[0] = *(const uint4*)(qbf + qbase + c * 32 + (half ? 8 : 0));
            qf[c].u[1] = *(const uint4*)(qbf + qbase + c * 32 + (half ? 24 : 16));
        }
    }
    int qseg[8], qfl[8];
    #pragma unroll
    for (int v = 0; v < 8; ++v) {
        int q = q0 + v + half * 8;
        qseg[v] = seg[b * SEQ + q];
        qfl[v]  = flg[b * SEQ + q];
    }
    float mrow[8], lrow[8];
    v8f acc[4];
    #pragma unroll
    for (int v = 0; v < 8; ++v) { mrow[v] = NEGI; lrow[v] = 0.f; }
    #pragma unroll
    for (int j = 0; j < 4; ++j) acc[j] = VZERO;

    for (int k0 = 0; k0 <= qhi; k0 += 32) {
        float sarr[2][8];
        #pragma unroll
        for (int t = 0; t < 2; ++t) {
            int key  = k0 + t * 16 + l16;
            int kseg = seg[b * SEQ + key];
            int kfl  = flg[b * SEQ + key];
            float raw8[8];
            if constexpr (MODE != 2) {
                size_t kbase = ((size_t)(b * SEQ + key)) * DMODEL + h * DKH;
                v8f cs = VZERO;
                #pragma unroll
                for (int c = 0; c < 2; ++c) {
                    Frag16 kf;
                    kf.u[0] = *(const uint4*)(kbf + kbase + c * 32 + half * 8);
                    kf.u[1] = *(const uint4*)(kbf + kbase + c * 32 + half * 8 + 16);
                    cs = __builtin_amdgcn_wmma_f32_16x16x32_bf16(
                        false, qf[c].v, false, kf.v, (short)0, cs, false, false);
                }
                #pragma unroll
                for (int v = 0; v < 8; ++v) raw8[v] = cs[v] * 0.125f;  // DK^-0.5
            } else {
                float aval = aproj[(size_t)(b * SEQ + key) * NH + h];
                #pragma unroll
                for (int v = 0; v < 8; ++v) raw8[v] = aval;
            }
            #pragma unroll
            for (int v = 0; v < 8; ++v) {
                int q = q0 + v + half * 8;
                bool ok; float bv;
                if constexpr (MODE == 1) {
                    ok = (key <= q) && (((kfl & 2) != 0) || (key == q));
                    bv = (float)(kseg - qseg[v]);
                } else {
                    bool eq = (kseg == qseg[v]) && ((qfl[v] & 1) != 0) && ((kfl & 1) != 0);
                    ok = (key <= q) && (eq || (key == q));
                    bv = (float)(key - q);
                }
                sarr[t][v] = ok ? (raw8[v] + bv * hs) : NEGI;
            }
        }
        // online softmax
        float alpha[8], mnew[8];
        #pragma unroll
        for (int v = 0; v < 8; ++v) {
            float rm = fmaxf(sarr[0][v], sarr[1][v]);
            #pragma unroll
            for (int m = 1; m < 16; m <<= 1) rm = fmaxf(rm, __shfl_xor(rm, m, 32));
            mnew[v]  = fmaxf(mrow[v], rm);
            alpha[v] = expf(mrow[v] - mnew[v]);
        }
        #pragma unroll
        for (int v = 0; v < 8; ++v) {
            float p0 = (sarr[0][v] <= -1e29f) ? 0.f : expf(sarr[0][v] - mnew[v]);
            float p1 = (sarr[1][v] <= -1e29f) ? 0.f : expf(sarr[1][v] - mnew[v]);
            sarr[0][v] = p0; sarr[1][v] = p1;
            float r = p0 + p1;
            #pragma unroll
            for (int m = 1; m < 16; m <<= 1) r += __shfl_xor(r, m, 32);
            lrow[v] = lrow[v] * alpha[v] + r;
            mrow[v] = mnew[v];
        }
        #pragma unroll
        for (int j = 0; j < 4; ++j)
            #pragma unroll
            for (int v = 0; v < 8; ++v) acc[j][v] *= alpha[v];

        // stage P in LDS (C layout -> A-fragment interleave), then PV WMMAs
        unsigned short* pb = Pb[wid];
        #pragma unroll
        for (int t = 0; t < 2; ++t)
            #pragma unroll
            for (int v = 0; v < 8; ++v)
                pb[(v + half * 8) * 40 + t * 16 + l16] = f2bf(sarr[t][v]);
        asm volatile("s_wait_dscnt 0" ::: "memory");
        Frag16 pf;
        pf.u[0] = *(const uint4*)&pb[l16 * 40 + (half ? 8 : 0)];
        pf.u[1] = *(const uint4*)&pb[l16 * 40 + (half ? 24 : 16)];
        #pragma unroll
        for (int j = 0; j < 4; ++j) {
            Frag16 vf;
            size_t vbase = ((size_t)((b * NH + h) * DKH + (j * 16 + l16))) * SEQ + k0;
            vf.u[0] = *(const uint4*)(vt + vbase + half * 8);
            vf.u[1] = *(const uint4*)(vt + vbase + half * 8 + 16);
            acc[j] = __builtin_amdgcn_wmma_f32_16x16x32_bf16(
                false, pf.v, false, vf.v, (short)0, acc[j], false, false);
        }
    }
    // epilogue: 1/l, head gate, bf16 store
    #pragma unroll
    for (int v = 0; v < 8; ++v) {
        int q = q0 + v + half * 8;
        float inv = 1.0f / lrow[v];
        float g = gate[(size_t)(b * SEQ + q) * NH + h];
        #pragma unroll
        for (int j = 0; j < 4; ++j)
            obf[(size_t)(b * SEQ + q) * DMODEL + h * DKH + j * 16 + l16] =
                f2bf(acc[j][v] * inv * g);
    }
}

// ---------------------------------------------------------------------------
extern "C" void kernel_launch(void* const* d_in, const int* in_sizes, int n_in,
                              void* d_out, int out_size, void* d_ws, size_t ws_size,
                              hipStream_t stream) {
    (void)in_sizes; (void)n_in; (void)out_size; (void)ws_size;
    const float* states = (const float*)d_in[0];
    const int*   ids    = (const int*)d_in[1];
    const float* lnw[3] = {(const float*)d_in[2],  (const float*)d_in[14], (const float*)d_in[23]};
    const float* lnb[3] = {(const float*)d_in[3],  (const float*)d_in[15], (const float*)d_in[24]};
    const float* qw[3]  = {(const float*)d_in[4],  nullptr,                (const float*)d_in[25]};
    const float* qb[3]  = {(const float*)d_in[5],  nullptr,                (const float*)d_in[26]};
    const float* kw[3]  = {(const float*)d_in[6],  nullptr,                (const float*)d_in[27]};
    const float* kb[3]  = {(const float*)d_in[7],  nullptr,                (const float*)d_in[28]};
    const float* vw[3]  = {(const float*)d_in[8],  (const float*)d_in[17], (const float*)d_in[29]};
    const float* vb[3]  = {(const float*)d_in[9],  (const float*)d_in[18], (const float*)d_in[30]};
    const float* gw[3]  = {(const float*)d_in[10], (const float*)d_in[19], (const float*)d_in[31]};
    const float* gb[3]  = {(const float*)d_in[11], (const float*)d_in[20], (const float*)d_in[32]};
    const float* ow[3]  = {(const float*)d_in[12], (const float*)d_in[21], (const float*)d_in[33]};
    const float* ob[3]  = {(const float*)d_in[13], (const float*)d_in[22], (const float*)d_in[34]};
    const float* aw1    = (const float*)d_in[16];

    char* ws = (char*)d_ws;
    size_t off = 0;
    auto alloc = [&](size_t bytes) -> void* {
        void* p = ws + off;
        off = (off + bytes + 255) & ~(size_t)255;
        return p;
    };
    unsigned short* lnbf  = (unsigned short*)alloc((size_t)NTOK * DMODEL * 2);
    unsigned short* qbfb  = (unsigned short*)alloc((size_t)NTOK * DMODEL * 2);
    unsigned short* kbfb  = (unsigned short*)alloc((size_t)NTOK * DMODEL * 2);
    unsigned short* vtb   = (unsigned short*)alloc((size_t)NTOK * DMODEL * 2);
    unsigned short* obfb  = (unsigned short*)alloc((size_t)NTOK * DMODEL * 2);
    unsigned short* wqb   = (unsigned short*)alloc((size_t)DMODEL * DMODEL * 2);
    unsigned short* wkb   = (unsigned short*)alloc((size_t)DMODEL * DMODEL * 2);
    unsigned short* wvb   = (unsigned short*)alloc((size_t)DMODEL * DMODEL * 2);
    unsigned short* wob   = (unsigned short*)alloc((size_t)DMODEL * DMODEL * 2);
    unsigned short* wgb   = (unsigned short*)alloc((size_t)NH * DMODEL * 2);
    unsigned short* wab   = (unsigned short*)alloc((size_t)NH * DMODEL * 2);
    float* gateb  = (float*)alloc((size_t)NTOK * NH * 4);
    float* aprojb = (float*)alloc((size_t)NTOK * NH * 4);
    int* segb = (int*)alloc((size_t)NTOK * 4);
    int* flgb = (int*)alloc((size_t)NTOK * 4);

    float* xf = (float*)d_out;   // residual stream lives in d_out
    hipMemcpyAsync(xf, states, (size_t)NTOK * DMODEL * sizeof(float),
                   hipMemcpyDeviceToDevice, stream);
    meta_kernel<<<BATCH, SEQ, 0, stream>>>(ids, segb, flgb);

    auto conv = [&](const float* src, unsigned short* dst, int n) {
        convert_kernel<<<(n + 255) / 256, 256, 0, stream>>>(src, dst, n);
    };

    dim3 gg(DMODEL / 64, NTOK / 128);
    const int attnGrid = (BATCH * NH * (SEQ / 16)) / 4;  // 4 waves per block
    const int hdotGrid = NTOK * NH / 8;

    for (int L = 0; L < 3; ++L) {
        ln_kernel<<<NTOK, 256, 0, stream>>>(xf, lnw[L], lnb[L], lnbf);
        conv(vw[L], wvb, DMODEL * DMODEL);
        conv(ow[L], wob, DMODEL * DMODEL);
        conv(gw[L], wgb, NH * DMODEL);
        gemm_kernel<<<gg, 256, 0, stream>>>(lnbf, wvb, vb[L], vtb, nullptr, 2);
        hdot_kernel<<<hdotGrid, 256, 0, stream>>>(lnbf, wgb, gb[L], gateb, 1);
        if (L == 1) {
            conv(aw1, wab, NH * DMODEL);
            hdot_kernel<<<hdotGrid, 256, 0, stream>>>(lnbf, wab, nullptr, aprojb, 0);
            attn_kernel<2><<<attnGrid, 128, 0, stream>>>(nullptr, nullptr, vtb, aprojb,
                                                         gateb, segb, flgb, obfb);
        } else {
            conv(qw[L], wqb, DMODEL * DMODEL);
            conv(kw[L], wkb, DMODEL * DMODEL);
            gemm_kernel<<<gg, 256, 0, stream>>>(lnbf, wqb, qb[L], qbfb, nullptr, 1);
            gemm_kernel<<<gg, 256, 0, stream>>>(lnbf, wkb, kb[L], kbfb, nullptr, 1);
            if (L == 0)
                attn_kernel<0><<<attnGrid, 128, 0, stream>>>(qbfb, kbfb, vtb, nullptr,
                                                             gateb, segb, flgb, obfb);
            else
                attn_kernel<1><<<attnGrid, 128, 0, stream>>>(qbfb, kbfb, vtb, nullptr,
                                                             gateb, segb, flgb, obfb);
        }
        gemm_kernel<<<gg, 256, 0, stream>>>(obfb, wob, ob[L], nullptr, xf, 3);
    }
}
